// SparseAttention_30528627540537
// MI455X (gfx1250) — matrix operand order repaired and checked by reading, not compiled, not run on hardware
//
#include <hip/hip_runtime.h>
#include <hip/hip_bf16.h>
#include <stdint.h>

// ---------------------------------------------------------------------------
// Sparse attention for MI455X (gfx1250): all GEMMs via v_wmma_f32_16x16x32_bf16
// B=2, N=2048, C=1024, H=16, hd=64, k_sparse=204
// ---------------------------------------------------------------------------

typedef __attribute__((ext_vector_type(16))) __bf16   bf16x16;
typedef __attribute__((ext_vector_type(8)))  float    f32x8;
typedef __attribute__((ext_vector_type(8)))  uint32_t u32x8;
typedef __attribute__((ext_vector_type(16))) uint16_t u16x16;

__device__ __forceinline__ uint16_t f2bf(float f) {
  uint32_t u = __float_as_uint(f);
  uint32_t r = 0x7fffu + ((u >> 16) & 1u);   // round-to-nearest-even
  return (uint16_t)((u + r) >> 16);
}
// monotonic float <-> sortable-uint mapping (for top-k threshold search)
__device__ __forceinline__ uint32_t sort_map(float f) {
  uint32_t x = __float_as_uint(f);
  return (x & 0x80000000u) ? ~x : (x | 0x80000000u);
}
__device__ __forceinline__ float sort_unmap(uint32_t u) {
  uint32_t x = (u & 0x80000000u) ? (u & 0x7fffffffu) : ~u;
  return __uint_as_float(x);
}

__device__ __forceinline__ f32x8 wmma_bf16(bf16x16 a, bf16x16 b, f32x8 c) {
  return __builtin_amdgcn_wmma_f32_16x16x32_bf16(false, a, false, b,
                                                 (short)0, c, false, false);
}

// ---------------- conversion kernels ----------------
__global__ void __launch_bounds__(256) cvt_kernel(const float* __restrict__ src,
                                                  uint16_t* __restrict__ dst, int n) {
  int i = blockIdx.x * 256 + threadIdx.x;
  if (i < n) dst[i] = f2bf(src[i]);
}

// dst[c*R + r] = bf16(src[r*C + c])  (transpose so B-fragments load contiguously)
__global__ void __launch_bounds__(256) cvtT_kernel(const float* __restrict__ src,
                                                   uint16_t* __restrict__ dst,
                                                   int R, int C) {
  int i = blockIdx.x * 256 + threadIdx.x;
  if (i < R * C) {
    int r = i / C, c = i % C;
    dst[c * R + r] = f2bf(src[i]);
  }
}

// ---------------- QKV projection: [4096,1024] @ [1024,3072] + bias ----------
// Output scattered into Q/K/V as bf16 [B,H,N,hd].
__global__ void __launch_bounds__(256) qkv_gemm_kernel(
    const uint16_t* __restrict__ xb, const uint16_t* __restrict__ wT,
    const float* __restrict__ bias,
    uint16_t* __restrict__ qb, uint16_t* __restrict__ kb,
    uint16_t* __restrict__ vb) {
  const int lane = threadIdx.x & 31;
  const int wave = threadIdx.x >> 5;
  const int tile = blockIdx.x * 8 + wave;          // 49152 tiles total
  const int tm = tile / 192, tn = tile % 192;
  const int row0 = tm * 16, col0 = tn * 16;
  const int nl = lane & 15;
  const int hi = (lane >> 4) & 1;                  // lane group
  const uint32_t* arow = (const uint32_t*)(xb + (size_t)(row0 + nl) * 1024);
  const uint32_t* brow = (const uint32_t*)(wT + (size_t)(col0 + nl) * 1024);
  f32x8 acc = {};
  for (int k0 = 0; k0 < 1024; k0 += 32) {
    const int ab = (k0 + hi * 16) >> 1;            // A: K = k0 + 16*hi + {0..15}
    const int bb = (k0 + hi * 16) >> 1;            // B: K = dkB + {0..15}
    u32x8 ar, br;
#pragma unroll
    for (int t = 0; t < 8; ++t) {
      ar[t] = arow[ab + t];
      br[t] = brow[bb + t];
    }
    __builtin_prefetch(arow + ab + 16, 0, 1);      // global_prefetch next chunk
    acc = wmma_bf16(__builtin_bit_cast(bf16x16, ar),
                    __builtin_bit_cast(bf16x16, br), acc);
  }
  const int c3 = col0 + nl;
  const float bv = bias[c3];
  const int which = c3 >> 10;                      // 0=q 1=k 2=v
  const int c = c3 & 1023;
  const int h = c >> 6, d = c & 63;
  uint16_t* outp = (which == 0) ? qb : (which == 1) ? kb : vb;
#pragma unroll
  for (int r = 0; r < 8; ++r) {
    const int row = row0 + hi * 8 + r;             // C/D: M = r + 8*hi
    const int bI = row >> 11, nI = row & 2047;
    outp[(size_t)((bI * 16 + h) * 2048 + nI) * 64 + d] = f2bf(acc[r] + bv);
  }
}

// ---------------- attention: scores -> exact top-204 -> softmax -> P@V ------
__global__ void __launch_bounds__(256) attn_kernel(
    const uint16_t* __restrict__ qb, const uint16_t* __restrict__ kbuf,
    const uint16_t* __restrict__ vbuf, uint16_t* __restrict__ ob) {
  extern __shared__ float smem[];
  float* S      = smem;                  // 16 x 2048 scores / probabilities
  float* OAp    = S + 16 * 2048;         // 8 x 16 x 64 per-wave partials
  float* rowInv = OAp + 8 * 16 * 64;     // 16 row 1/sum
  const int lane = threadIdx.x & 31;
  const int wave = threadIdx.x >> 5;
  const int b = blockIdx.z, h = blockIdx.y;
  const int row0 = blockIdx.x * 16;
  const int bh = b * 16 + h;
  const uint16_t* Q = qb + (size_t)bh * 2048 * 64;
  const uint16_t* K = kbuf + (size_t)bh * 2048 * 64;
  const uint16_t* V = vbuf + (size_t)bh * 2048 * 64;
  const int nl = lane & 15;
  const int hi = (lane >> 4) & 1;
  const int dkA = hi * 8, dkB = hi * 16;

  // Q A-fragments for hd-chunks [0,32) and [32,64), held for whole kernel
  const uint32_t* qrow = (const uint32_t*)(Q + (size_t)(row0 + nl) * 64);
  u32x8 q0r, q1r;
#pragma unroll
  for (int t = 0; t < 8; ++t) {
    q0r[t] = qrow[(dkB >> 1) + t];
    q1r[t] = qrow[((32 + dkB) >> 1) + t];
  }
  const bf16x16 q0 = __builtin_bit_cast(bf16x16, q0r);
  const bf16x16 q1 = __builtin_bit_cast(bf16x16, q1r);

  // -------- phase 1: S = (Q K^T) * hd^-0.5, 128 col-tiles over 8 waves
  for (int ct = wave; ct < 128; ct += 8) {
    const int col0 = ct * 16;
    const uint32_t* krow = (const uint32_t*)(K + (size_t)(col0 + nl) * 64);
    u32x8 b0r, b1r;
#pragma unroll
    for (int t = 0; t < 8; ++t) {
      b0r[t] = krow[(dkB >> 1) + t];
      b1r[t] = krow[((32 + dkB) >> 1) + t];
    }
    f32x8 acc = {};
    acc = wmma_bf16(q0, __builtin_bit_cast(bf16x16, b0r), acc);
    acc = wmma_bf16(q1, __builtin_bit_cast(bf16x16, b1r), acc);
#pragma unroll
    for (int r = 0; r < 8; ++r)
      S[(hi * 8 + r) * 2048 + col0 + nl] = acc[r] * 0.125f;
  }
  __syncthreads();

  // -------- phase 2: per-row exact top-204 threshold + softmax
  // wave w owns rows w and w+8; each lane caches 64 of 2048 entries
  for (int rr = 0; rr < 2; ++rr) {
    const int row = wave + rr * 8;
    uint32_t u[64];
#pragma unroll
    for (int i = 0; i < 64; ++i)
      u[i] = sort_map(S[row * 2048 + lane + 32 * i]);
    uint32_t mu = 0;
#pragma unroll
    for (int i = 0; i < 64; ++i) mu = (u[i] > mu) ? u[i] : mu;
    for (int off = 16; off > 0; off >>= 1) {
      uint32_t o = __shfl_xor(mu, off, 32);
      mu = (o > mu) ? o : mu;
    }
    const float rmax = sort_unmap(mu);
    // binary search over sortable bits: largest thr with count(u>=thr) >= 204
    uint32_t thr = 0;
    for (int bit = 31; bit >= 0; --bit) {
      const uint32_t cand = thr | (1u << bit);
      int cnt = 0;
#pragma unroll
      for (int i = 0; i < 64; ++i) cnt += (u[i] >= cand) ? 1 : 0;
      for (int off = 16; off > 0; off >>= 1) cnt += __shfl_xor(cnt, off, 32);
      if (cnt >= 204) thr = cand;
    }
    float sum = 0.f;
#pragma unroll
    for (int i = 0; i < 64; ++i) {
      float p = 0.f;
      if (u[i] >= thr) p = __expf(sort_unmap(u[i]) - rmax);
      sum += p;
      S[row * 2048 + lane + 32 * i] = p;   // unselected -> exactly 0
    }
    for (int off = 16; off > 0; off >>= 1) sum += __shfl_xor(sum, off, 32);
    if (lane == 0) rowInv[row] = 1.0f / sum;
  }
  __syncthreads();

  // -------- phase 3: O[16,64] = P[16,2048] @ V[2048,64]
  // wave w handles K-chunks {w, w+8, ...}; deterministic LDS reduction after
  f32x8 acc0 = {}, acc1 = {}, acc2 = {}, acc3 = {};
  const float invm = rowInv[nl];           // A row of this lane = nl
  for (int kc = wave; kc < 64; kc += 8) {
    const int k0 = kc * 32;
    u16x16 ar;
#pragma unroll
    for (int j = 0; j < 16; ++j) {
      const int kk = k0 + dkB + j;
      ar[j] = f2bf(S[nl * 2048 + kk] * invm);    // fold 1/sum into P
    }
    const bf16x16 a = __builtin_bit_cast(bf16x16, ar);
#pragma unroll
    for (int t = 0; t < 4; ++t) {
      u16x16 br;
#pragma unroll
      for (int j = 0; j < 16; ++j)
        br[j] = V[(size_t)(k0 + dkB + j) * 64 + t * 16 + nl];
      const bf16x16 bm = __builtin_bit_cast(bf16x16, br);
      f32x8& acc = (t == 0) ? acc0 : (t == 1) ? acc1 : (t == 2) ? acc2 : acc3;
      acc = wmma_bf16(a, bm, acc);
    }
  }
  float* myOA = OAp + wave * 16 * 64;
#pragma unroll
  for (int r = 0; r < 8; ++r) {
    myOA[(hi * 8 + r) * 64 +  0 + nl] = acc0[r];
    myOA[(hi * 8 + r) * 64 + 16 + nl] = acc1[r];
    myOA[(hi * 8 + r) * 64 + 32 + nl] = acc2[r];
    myOA[(hi * 8 + r) * 64 + 48 + nl] = acc3[r];
  }
  __syncthreads();
  // reduce the 8 wave partials and emit bf16 [B,N,C] (C index = h*64 + d)
  for (int i = threadIdx.x; i < 16 * 64; i += 256) {
    float s = 0.f;
#pragma unroll
    for (int w = 0; w < 8; ++w) s += OAp[w * 1024 + i];
    const int rr = i >> 6, d = i & 63;
    ob[(size_t)(b * 2048 + row0 + rr) * 1024 + h * 64 + d] = f2bf(s);
  }
}

// ---------------- output projection: [4096,1024] @ [1024,1024] + bias -------
__global__ void __launch_bounds__(256) out_gemm_kernel(
    const uint16_t* __restrict__ ab16, const uint16_t* __restrict__ wT,
    const float* __restrict__ bias, float* __restrict__ out) {
  const int lane = threadIdx.x & 31;
  const int wave = threadIdx.x >> 5;
  const int tile = blockIdx.x * 8 + wave;          // 16384 tiles total
  const int tm = tile >> 6, tn = tile & 63;
  const int row0 = tm * 16, col0 = tn * 16;
  const int nl = lane & 15;
  const int hi = (lane >> 4) & 1;
  const uint32_t* arow = (const uint32_t*)(ab16 + (size_t)(row0 + nl) * 1024);
  const uint32_t* brow = (const uint32_t*)(wT + (size_t)(col0 + nl) * 1024);
  f32x8 acc = {};
  for (int k0 = 0; k0 < 1024; k0 += 32) {
    const int ab = (k0 + hi * 16) >> 1;
    const int bb = (k0 + hi * 16) >> 1;
    u32x8 ar, br;
#pragma unroll
    for (int t = 0; t < 8; ++t) {
      ar[t] = arow[ab + t];
      br[t] = brow[bb + t];
    }
    acc = wmma_bf16(__builtin_bit_cast(bf16x16, ar),
                    __builtin_bit_cast(bf16x16, br), acc);
  }
  const float bv = bias[col0 + nl];
#pragma unroll
  for (int r = 0; r < 8; ++r)
    out[(size_t)(row0 + hi * 8 + r) * 1024 + col0 + nl] = acc[r] + bv;
}

// ---------------------------------------------------------------------------
extern "C" void kernel_launch(void* const* d_in, const int* in_sizes, int n_in,
                              void* d_out, int out_size, void* d_ws, size_t ws_size,
                              hipStream_t stream) {
  (void)in_sizes; (void)n_in; (void)out_size; (void)ws_size;
  const float* x    = (const float*)d_in[0];   // [2,2048,1024]
  const float* wqkv = (const float*)d_in[1];   // [1024,3072]
  const float* bqkv = (const float*)d_in[2];   // [3072]
  const float* wout = (const float*)d_in[3];   // [1024,1024]
  const float* bout = (const float*)d_in[4];   // [1024]
  float* out = (float*)d_out;                  // [2,2048,1024] f32

  // workspace layout (bf16 elements), total 48 MiB
  uint16_t* ws    = (uint16_t*)d_ws;
  uint16_t* x_bf  = ws;                        // 4096*1024
  uint16_t* wqkvT = x_bf  + 4096 * 1024;       // 3072*1024 (transposed)
  uint16_t* woutT = wqkvT + 3072 * 1024;       // 1024*1024 (transposed)
  uint16_t* q_bf  = woutT + 1024 * 1024;       // [B,H,N,hd] = 4096*1024
  uint16_t* k_bf  = q_bf  + 4096 * 1024;
  uint16_t* v_bf  = k_bf  + 4096 * 1024;
  uint16_t* a_bf  = v_bf  + 4096 * 1024;       // [B,N,C]    = 4096*1024

  cvt_kernel<<<(4096 * 1024) / 256, 256, 0, stream>>>(x, x_bf, 4096 * 1024);
  cvtT_kernel<<<(3072 * 1024) / 256, 256, 0, stream>>>(wqkv, wqkvT, 1024, 3072);
  cvtT_kernel<<<(1024 * 1024) / 256, 256, 0, stream>>>(wout, woutT, 1024, 1024);

  qkv_gemm_kernel<<<6144, 256, 0, stream>>>(x_bf, wqkvT, bqkv, q_bf, k_bf, v_bf);

  dim3 ag(128, 16, 2);  // (row-block, head, batch)
  size_t shmem = (size_t)(16 * 2048 + 8 * 16 * 64 + 16) * sizeof(float); // ~160 KB
  attn_kernel<<<ag, 256, shmem, stream>>>(q_bf, k_bf, v_bf, a_bf);

  out_gemm_kernel<<<2048, 256, 0, stream>>>(a_bf, woutT, bout, out);
}
